// GCN_23330262352501
// MI455X (gfx1250) — matrix operand order repaired and checked by reading, not compiled, not run on hardware
//
#include <hip/hip_runtime.h>
#include <stdint.h>

// ---------------------------------------------------------------------------
// 2-layer GCN for MI455X (gfx1250, wave32).
//   GEMM phase:  v_wmma_f32_16x16x32_bf16 (bf16 A/B, f32 accumulate),
//                B slab staged in LDS (conflict-free ds_load feeds).
//   Aggregation: one wave per edge, float4 gathers (L2-resident: h = 51.2MB
//                fits in the 192MB L2), global_atomic_add_f32 scatter.
// ---------------------------------------------------------------------------

typedef __attribute__((ext_vector_type(16))) __bf16 v16bf;
typedef __attribute__((ext_vector_type(8)))  __bf16 v8bf;
typedef __attribute__((ext_vector_type(8)))  float  v8f;

#define FDIM 256
#define LDS_STRIDE 264   // 256 + 8 bf16 pad: 528B row stride -> bank stride 4

__device__ __forceinline__ __bf16 f2bf(float f) { return (__bf16)f; }

// ---- degree with self-loop, then dis = rsqrt(deg) -------------------------
__global__ void k_deg_init(float* deg, int n) {
  int i = blockIdx.x * blockDim.x + threadIdx.x;
  if (i < n) deg[i] = 1.0f;                       // self-loop contribution
}
__global__ void k_deg_acc(const int* __restrict__ dst, float* deg, int e) {
  int i = blockIdx.x * blockDim.x + threadIdx.x;
  if (i < e) atomicAdd(&deg[dst[i]], 1.0f);
}
__global__ void k_deg_rsqrt(float* deg, int n) {
  int i = blockIdx.x * blockDim.x + threadIdx.x;
  if (i < n) deg[i] = rsqrtf(deg[i]);             // deg >= 1 always
}

// ---- Wt[n][k] = bf16(W[k][n]) : B-fragment loads become contiguous --------
__global__ void k_prep_w(const float* __restrict__ W, __bf16* __restrict__ Wt) {
  int t = blockIdx.x * blockDim.x + threadIdx.x;  // FDIM*FDIM threads
  int k = t >> 8, n = t & 255;
  Wt[n * FDIM + k] = f2bf(W[k * FDIM + n]);
}

// ---- H[M,256] = bf16(X[M,256]) @ W  via WMMA ------------------------------
// block = 256 threads (8 waves). Each wave: 16(M) x 64(N) tile, K-loop of 8.
// The block's 64-column W slab (32KB bf16) is staged in LDS once and shared
// by all 8 waves; inner loop feeds B from conflict-free ds_load_b128 pairs.
// grid = (ceil(M/128), FDIM/64)
__global__ void __launch_bounds__(256)
k_gemm(const float* __restrict__ X, const __bf16* __restrict__ Wt,
       float* __restrict__ H, int M) {
  __shared__ __bf16 lds_w[64 * LDS_STRIDE];       // 33,792 bytes

  const int wave  = threadIdx.x >> 5;
  const int lane  = threadIdx.x & 31;
  const int hlane = lane & 15;
  const int hi    = lane >> 4;                    // 0 or 1 (half-wave)
  const int m0 = blockIdx.x * 128 + wave * 16;
  const int n0 = blockIdx.y * 64;

  // ---- cooperative stage: 64 cols x 512B = 2048 x 16B chunks, 8/thread ----
  {
    const uint4* gsrc = (const uint4*)(Wt + (size_t)n0 * FDIM); // contiguous 32KB
    char* lbase = (char*)lds_w;
#pragma unroll
    for (int it = 0; it < 8; ++it) {
      int c = it * 256 + threadIdx.x;             // chunk id 0..2047
      uint4 v = gsrc[c];
      *(uint4*)(lbase + (c >> 5) * (LDS_STRIDE * 2) + (c & 31) * 16) = v;
    }
  }
  __syncthreads();
  if (m0 >= M) return;                            // wave-uniform: EXEC stays full

  // A layout (16-bit 16x32): lane<16 holds row M=lane, K = {0..7, 16..23};
  // lane>=16 holds row M=lane-16, K = {8..15, 24..31}.
  const float* xrow = X + (size_t)(m0 + hlane) * FDIM + hi * 8;

  v8f acc[4] = {};
  for (int k0 = 0; k0 < FDIM; k0 += 32) {
    const float4* xp = (const float4*)(xrow + k0);
    float4 a0 = xp[0], a1 = xp[1];                // K = k0 + hi*8 + 0..7
    float4 a2 = xp[4], a3 = xp[5];                // K = k0 + hi*8 + 16..23
    v16bf a;
    a[0]  = f2bf(a0.x); a[1]  = f2bf(a0.y); a[2]  = f2bf(a0.z); a[3]  = f2bf(a0.w);
    a[4]  = f2bf(a1.x); a[5]  = f2bf(a1.y); a[6]  = f2bf(a1.z); a[7]  = f2bf(a1.w);
    a[8]  = f2bf(a2.x); a[9]  = f2bf(a2.y); a[10] = f2bf(a2.z); a[11] = f2bf(a2.w);
    a[12] = f2bf(a3.x); a[13] = f2bf(a3.y); a[14] = f2bf(a3.z); a[15] = f2bf(a3.w);

#pragma unroll
    for (int t = 0; t < 4; ++t) {
      // B layout (32x16 bf16): lane<16 -> col=lane, K=k0+{0..7, 16..23};
      // lane>=16 -> col=lane-16, K=k0+{8..15, 24..31}.
      // LDS row stride 528B => lane n hits bank group 4n (conflict-free).
      const __bf16* bp = lds_w + (t * 16 + hlane) * LDS_STRIDE + k0 + hi * 8;
      v8bf blo = *(const v8bf*)bp;                // 16B aligned
      v8bf bhi = *(const v8bf*)(bp + 16);
      v16bf b = __builtin_shufflevector(blo, bhi,
                                        0, 1, 2, 3, 4, 5, 6, 7,
                                        8, 9, 10, 11, 12, 13, 14, 15);
      acc[t] = __builtin_amdgcn_wmma_f32_16x16x32_bf16(
          /*neg_a=*/false, a, /*neg_b=*/false, b,
          /*c_mod=*/(short)0, acc[t], /*reuse_a=*/false, /*reuse_b=*/false);
    }
  }

  // D layout (32-bit 16x16): VGPR j -> M = m0 + j + 8*hi, N = n0 + t*16 + hlane
  const int mbase = m0 + hi * 8;
#pragma unroll
  for (int t = 0; t < 4; ++t) {
    float* o = H + (size_t)mbase * FDIM + n0 + t * 16 + hlane;
#pragma unroll
    for (int j = 0; j < 8; ++j) o[(size_t)j * FDIM] = acc[t][j];
  }
}

// ---- out = h * dis^2 (self-loop) + bias -----------------------------------
__global__ void k_self_bias(const float* __restrict__ H, const float* __restrict__ dis,
                            const float* __restrict__ bias, float* __restrict__ out,
                            int n) {
  size_t t = (size_t)(blockIdx.x * blockDim.x + threadIdx.x) * 4;
  if (t >= (size_t)n * FDIM) return;
  int node = (int)(t >> 8), c = (int)(t & 255);
  float s = dis[node]; s = s * s;
  float4 h = *(const float4*)(H + t);
  float4 b = *(const float4*)(bias + c);
  float4 o = make_float4(h.x * s + b.x, h.y * s + b.y, h.z * s + b.z, h.w * s + b.w);
  *(float4*)(out + t) = o;
}

// ---- edge scatter: out[dst] += h[src] * dis[src]*dis[dst] -----------------
// one wave (32 lanes) per edge; each lane covers 8 contiguous channels.
__global__ void __launch_bounds__(256)
k_edge(const float* __restrict__ H, const int* __restrict__ src,
       const int* __restrict__ dstv, const float* __restrict__ dis,
       float* __restrict__ out, int e) {
  int edge = blockIdx.x * 8 + (threadIdx.x >> 5);
  if (edge >= e) return;
  int lane = threadIdx.x & 31;
  int s = src[edge], d = dstv[edge];
  float nrm = dis[s] * dis[d];
  const float4* hp = (const float4*)(H + (size_t)s * FDIM) + lane * 2;
  float4 v0 = hp[0], v1 = hp[1];
  float* op = out + (size_t)d * FDIM + lane * 8;
  atomicAdd(op + 0, v0.x * nrm); atomicAdd(op + 1, v0.y * nrm);
  atomicAdd(op + 2, v0.z * nrm); atomicAdd(op + 3, v0.w * nrm);
  atomicAdd(op + 4, v1.x * nrm); atomicAdd(op + 5, v1.y * nrm);
  atomicAdd(op + 6, v1.z * nrm); atomicAdd(op + 7, v1.w * nrm);
}

// ---- per-channel PReLU, in place ------------------------------------------
__global__ void k_prelu(float* out, const float* __restrict__ a, int n) {
  size_t t = (size_t)(blockIdx.x * blockDim.x + threadIdx.x) * 4;
  if (t >= (size_t)n * FDIM) return;
  int c = (int)(t & 255);
  float4 v = *(float4*)(out + t);
  float4 aa = *(const float4*)(a + c);
  v.x = v.x >= 0.0f ? v.x : aa.x * v.x;
  v.y = v.y >= 0.0f ? v.y : aa.y * v.y;
  v.z = v.z >= 0.0f ? v.z : aa.z * v.z;
  v.w = v.w >= 0.0f ? v.w : aa.w * v.w;
  *(float4*)(out + t) = v;
}

// ---------------------------------------------------------------------------
extern "C" void kernel_launch(void* const* d_in, const int* in_sizes, int n_in,
                              void* d_out, int out_size, void* d_ws, size_t ws_size,
                              hipStream_t stream) {
  (void)n_in; (void)out_size; (void)ws_size;
  const float* x  = (const float*)d_in[0];
  const int*   ei = (const int*)d_in[1];
  const float* W1 = (const float*)d_in[2];
  const float* b1 = (const float*)d_in[3];
  const float* W2 = (const float*)d_in[4];
  const float* b2 = (const float*)d_in[5];
  const float* pa = (const float*)d_in[6];

  const int N = in_sizes[0] / FDIM;     // 50000
  const int E = in_sizes[1] / 2;        // 1.6M
  const int* src = ei;
  const int* dst = ei + E;

  // workspace layout (needs ~51.7 MB):
  //   [0x000000) dis        : N floats (degree, then rsqrt in place)
  //   [0x040000) Wt1 (bf16) : 128 KB
  //   [0x060000) Wt2 (bf16) : 128 KB
  //   [0x080000) h          : N*256 floats (per-layer GEMM output)
  char*   ws  = (char*)d_ws;
  float*  dis = (float*)ws;
  __bf16* Wt1 = (__bf16*)(ws + 0x40000);
  __bf16* Wt2 = (__bf16*)(ws + 0x60000);
  float*  h   = (float*)(ws + 0x80000);
  float*  out = (float*)d_out;

  const int thr = 256;
  const int q4  = (N * FDIM) / 4;       // float4 elements per feature map

  k_deg_init <<<(N + thr - 1) / thr, thr, 0, stream>>>(dis, N);
  k_deg_acc  <<<(E + thr - 1) / thr, thr, 0, stream>>>(dst, dis, E);
  k_deg_rsqrt<<<(N + thr - 1) / thr, thr, 0, stream>>>(dis, N);
  k_prep_w   <<<(FDIM * FDIM) / thr, thr, 0, stream>>>(W1, Wt1);
  k_prep_w   <<<(FDIM * FDIM) / thr, thr, 0, stream>>>(W2, Wt2);

  dim3 gg((N + 127) / 128, FDIM / 64);

  // ---- layer 1: h = x @ W1 ; z1 (in d_out) = aggregate(h) + b1 ----
  k_gemm     <<<gg, 256, 0, stream>>>(x, Wt1, h, N);
  k_self_bias<<<(q4 + thr - 1) / thr, thr, 0, stream>>>(h, dis, b1, out, N);
  k_edge     <<<(E + 7) / 8, 256, 0, stream>>>(h, src, dst, dis, out, E);

  // ---- layer 2: h = z1 @ W2 ; out = aggregate(h) + b2 ; PReLU ----
  k_gemm     <<<gg, 256, 0, stream>>>(out, Wt2, h, N);   // reads z1 before...
  k_self_bias<<<(q4 + thr - 1) / thr, thr, 0, stream>>>(h, dis, b2, out, N); // ...overwrite
  k_edge     <<<(E + 7) / 8, 256, 0, stream>>>(h, src, dst, dis, out, E);
  k_prelu    <<<(q4 + thr - 1) / thr, thr, 0, stream>>>(out, pa, N);
}
